// LinearAttention_25331717112434
// MI455X (gfx1250) — matrix operand order repaired and checked by reading, not compiled, not run on hardware
//
#include <hip/hip_runtime.h>
#include <hip/hip_bf16.h>

// ---------------------------------------------------------------------------
// Linear attention on MI455X (gfx1250): everything as bf16 WMMA GEMMs with
// double-buffered LDS fed by GLOBAL_LOAD_ASYNC_TO_LDS_B128 (ASYNCcnt path).
//   Xb = bf16(x); Wb = bf16(W); projT = bf16(proj)^T
//   Q = Xb Wq^T ; K = Xb Wk^T ; V = Xb Wv^T          (M=16384,N=1024,K=1024)
//   Qp = elu1(Q projT^T) ; Kp = elu1(K projT^T)      (M=16384,N=256, K=1024)
//   z  = 1/(rowsum(Qp)+eps)
//   kvT[b] = V[b]^T Kp[b]                            (M=1024, N=256, K=4096)
//   out[b] = (Qp[b] kvT[b]^T) * z                    (M=4096, N=1024,K=256) fp32
// ---------------------------------------------------------------------------

typedef __attribute__((ext_vector_type(16))) __bf16 v16bf;
typedef __attribute__((ext_vector_type(8)))  float  v8f;

union FragU { v16bf v; uint4 q[2]; };

#define BM 128
#define BN 128
#define BK 32
#define LDSS 40   // (BK + 8) bf16 -> 80-byte row stride: 16B aligned, bank-staggered

enum { EPI_BF16 = 0, EPI_ELU1 = 1, EPI_ZOUT = 2 };

// Async 16-byte global -> LDS copy (no VGPR round trip, tracked by ASYNCcnt).
__device__ __forceinline__ void async_copy_b128(void* lds_dst, const void* gsrc)
{
    const unsigned           laddr = (unsigned)(unsigned long long)lds_dst;
    const unsigned long long gaddr = (unsigned long long)gsrc;
    asm volatile("global_load_async_to_lds_b128 %0, %1, off"
                 :: "v"(laddr), "v"(gaddr) : "memory");
}

__device__ __forceinline__ void wait_async0()
{
    asm volatile("s_wait_asynccnt 0x0" ::: "memory");
}

template<bool A_TRANS, bool B_KN, int EPI>
__global__ __launch_bounds__(256)
void gemm_wmma(const __bf16* __restrict__ A, int lda, long strideA,
               const __bf16* __restrict__ Bm, int ldb, long strideB,
               void* __restrict__ Cv, int ldc, long strideC,
               const float* __restrict__ zvec, long strideZ,
               int K)
{
    __shared__ __bf16 As[2][BM][LDSS];
    __shared__ __bf16 Bs[2][BN][LDSS];

    const int bz = blockIdx.z;
    A    += (long)bz * strideA;
    Bm   += (long)bz * strideB;
    zvec += (long)bz * strideZ;

    const int m0    = blockIdx.y * BM;
    const int n0    = blockIdx.x * BN;
    const int tid   = threadIdx.x;
    const int wave  = tid >> 5;
    const int lane  = tid & 31;
    const int wm    = (wave >> 2) * 64;   // wave M offset: 0 / 64
    const int wn    = (wave & 3)  * 32;   // wave N offset: 0/32/64/96
    const int l16   = lane & 15;
    const int khalf = lane >> 4;

    v8f acc[4][2];
#pragma unroll
    for (int i = 0; i < 4; ++i)
#pragma unroll
        for (int j = 0; j < 2; ++j)
#pragma unroll
            for (int r = 0; r < 8; ++r) acc[i][j][r] = 0.0f;

    // Stage one BK-deep K tile of A and B into LDS buffer `buf`.
    auto stage = [&](int buf, int k0) {
#pragma unroll
        for (int it = 0; it < 2; ++it) {
            const int c = tid + it * 256;          // 512 x 16B chunks per operand
            if (!A_TRANS) {
                const int r  = c >> 2;
                const int kk = (c & 3) << 3;
                async_copy_b128(&As[buf][r][kk],
                                A + (long)(m0 + r) * lda + (k0 + kk));
            } else {
                // A[m][k] = src[(k0+k)*lda + m0+m]; transpose during staging
                const int kk = c >> 4;
                const int m8 = (c & 15) << 3;
                __builtin_prefetch(A + (long)(k0 + BK + kk) * lda + (m0 + m8), 0, 3);
                const uint4 vv = *(const uint4*)(A + (long)(k0 + kk) * lda + (m0 + m8));
                const __bf16* pv = (const __bf16*)&vv;
#pragma unroll
                for (int i2 = 0; i2 < 8; ++i2) As[buf][m8 + i2][kk] = pv[i2];
            }
            if (!B_KN) {
                const int r  = c >> 2;
                const int kk = (c & 3) << 3;
                async_copy_b128(&Bs[buf][r][kk],
                                Bm + (long)(n0 + r) * ldb + (k0 + kk));
            } else {
                // src is [K][N]: transpose while staging
                const int kk = c >> 4;
                const int n8 = (c & 15) << 3;
                __builtin_prefetch(Bm + (long)(k0 + BK + kk) * ldb + (n0 + n8), 0, 3);
                const uint4 vv = *(const uint4*)(Bm + (long)(k0 + kk) * ldb + (n0 + n8));
                const __bf16* pv = (const __bf16*)&vv;
#pragma unroll
                for (int i2 = 0; i2 < 8; ++i2) Bs[buf][n8 + i2][kk] = pv[i2];
            }
        }
    };

    const int ktiles = K / BK;
    stage(0, 0);

    for (int kt = 0; kt < ktiles; ++kt) {
        const int buf = kt & 1;
        wait_async0();        // buffer `buf` contents have landed in LDS
        __syncthreads();      // all waves' stage of `buf` done; `buf^1` free

        if (kt + 1 < ktiles)
            stage(buf ^ 1, (kt + 1) * BK);   // overlap copy with compute below

        // ---- load fragments per CDNA5 16-bit WMMA layouts
        v16bf afrag[4];
        v16bf bfrag[2];
#pragma unroll
        for (int i = 0; i < 4; ++i) {
            // A 16x32: lane l<16 row M=l holds K 0..7 & 16..23; l>=16: 8..15 & 24..31
            const __bf16* rowp = &As[buf][wm + i * 16 + l16][0];
            FragU f;
            f.q[0] = *(const uint4*)(rowp + khalf * 8);
            f.q[1] = *(const uint4*)(rowp + 16 + khalf * 8);
            afrag[i] = f.v;
        }
#pragma unroll
        for (int j = 0; j < 2; ++j) {
            // B 32x16: lane l<16 col N=l holds K 0..7 & 16..23; l>=16: 8..15 & 24..31
            const __bf16* colp = &Bs[buf][wn + j * 16 + l16][0];
            FragU f;
            f.q[0] = *(const uint4*)(colp + khalf * 8);
            f.q[1] = *(const uint4*)(colp + 16 + khalf * 8);
            bfrag[j] = f.v;
        }

#pragma unroll
        for (int i = 0; i < 4; ++i)
#pragma unroll
            for (int j = 0; j < 2; ++j)
                acc[i][j] = __builtin_amdgcn_wmma_f32_16x16x32_bf16(
                    false, afrag[i], false, bfrag[j],
                    (short)0, acc[i][j], false, false);
        // No trailing barrier: buffer `buf` is only overwritten at kt+2, and the
        // top-of-loop barrier at kt+1 already orders all reads before that write.
    }

    // ---- epilogue: C/D layout -> VGPR r holds (M = r + 8*khalf, N = l16)
#pragma unroll
    for (int i = 0; i < 4; ++i) {
#pragma unroll
        for (int j = 0; j < 2; ++j) {
            const int col = n0 + wn + j * 16 + l16;
            const int mb  = m0 + wm + i * 16 + khalf * 8;
#pragma unroll
            for (int r = 0; r < 8; ++r) {
                float val = acc[i][j][r];
                const long row = mb + r;
                if (EPI == EPI_ELU1)
                    val = (val > 0.0f) ? (val + 1.0f) : __expf(val);
                if (EPI == EPI_ZOUT) {
                    float* C = (float*)Cv + (long)bz * strideC;
                    C[row * (long)ldc + col] = val * zvec[row];
                } else {
                    __bf16* C = (__bf16*)Cv + (long)bz * strideC;
                    C[row * (long)ldc + col] = (__bf16)val;
                }
            }
        }
    }
}

// ---------------------------------------------------------------------------
struct __align__(8) bf4 { __bf16 x, y, z, w; };

__global__ void cvt_f32_bf16(const float* __restrict__ in, __bf16* __restrict__ out, long n)
{
    const long i = ((long)blockIdx.x * blockDim.x + threadIdx.x) * 4;
    if (i >= n) return;                 // n is a multiple of 4 by construction
    float4 f = *(const float4*)(in + i);
    bf4 o;
    o.x = (__bf16)f.x; o.y = (__bf16)f.y; o.z = (__bf16)f.z; o.w = (__bf16)f.w;
    *(bf4*)(out + i) = o;
}

__global__ void transpose_cvt(const float* __restrict__ in, __bf16* __restrict__ out,
                              int rows, int cols)
{
    const int idx = blockIdx.x * blockDim.x + threadIdx.x;
    const int r = idx / cols, c = idx % cols;
    out[(long)c * rows + r] = (__bf16)in[(long)r * cols + c];
}

__global__ void rowsum_inv(const __bf16* __restrict__ Qp, float* __restrict__ z, int P)
{
    const int row  = blockIdx.x * 8 + (threadIdx.x >> 5);
    const int lane = threadIdx.x & 31;
    const __bf16* p = Qp + (long)row * P;
    float s = 0.0f;
    for (int j = lane; j < P; j += 32) s += (float)p[j];
#pragma unroll
    for (int off = 16; off > 0; off >>= 1) s += __shfl_xor(s, off, 32);
    if (lane == 0) z[row] = 1.0f / (s + 1e-8f);
}

// ---------------------------------------------------------------------------
extern "C" void kernel_launch(void* const* d_in, const int* in_sizes, int n_in,
                              void* d_out, int out_size, void* d_ws, size_t ws_size,
                              hipStream_t stream)
{
    (void)in_sizes; (void)n_in; (void)out_size; (void)ws_size;
    constexpr int  Bb = 4, Ll = 4096, Dd = 1024, Pp = 256;
    constexpr long M  = (long)Bb * Ll;          // 16384

    const float* x    = (const float*)d_in[0];
    const float* Wq   = (const float*)d_in[1];
    const float* Wk   = (const float*)d_in[2];
    const float* Wv   = (const float*)d_in[3];
    const float* proj = (const float*)d_in[4];

    char*  ws  = (char*)d_ws;
    size_t off = 0;
    auto alloc = [&](size_t bytes) -> void* {
        void* p = ws + off;
        off += (bytes + 255) & ~(size_t)255;
        return p;
    };
    __bf16* Xb    = (__bf16*)alloc(M * Dd * 2);
    __bf16* Wqb   = (__bf16*)alloc((size_t)Dd * Dd * 2);
    __bf16* Wkb   = (__bf16*)alloc((size_t)Dd * Dd * 2);
    __bf16* Wvb   = (__bf16*)alloc((size_t)Dd * Dd * 2);
    __bf16* projT = (__bf16*)alloc((size_t)Pp * Dd * 2);
    __bf16* Qb    = (__bf16*)alloc(M * Dd * 2);
    __bf16* Kb    = (__bf16*)alloc(M * Dd * 2);
    __bf16* Vb    = (__bf16*)alloc(M * Dd * 2);
    __bf16* Qp    = (__bf16*)alloc(M * Pp * 2);
    __bf16* Kp    = (__bf16*)alloc(M * Pp * 2);
    float*  zv    = (float*) alloc(M * 4);
    __bf16* kvT   = (__bf16*)alloc((size_t)Bb * Dd * Pp * 2);

    // 1) conversions
    cvt_f32_bf16<<<(int)(M * Dd / 4 / 256), 256, 0, stream>>>(x,  Xb,  M * Dd);
    cvt_f32_bf16<<<Dd * Dd / 4 / 256, 256, 0, stream>>>(Wq, Wqb, (long)Dd * Dd);
    cvt_f32_bf16<<<Dd * Dd / 4 / 256, 256, 0, stream>>>(Wk, Wkb, (long)Dd * Dd);
    cvt_f32_bf16<<<Dd * Dd / 4 / 256, 256, 0, stream>>>(Wv, Wvb, (long)Dd * Dd);
    transpose_cvt<<<Dd * Pp / 256, 256, 0, stream>>>(proj, projT, Dd, Pp);

    // 2) Q/K/V = Xb @ W^T  (W already [N][K] row-major)
    dim3 gQKV(Dd / BN, (int)(M / BM), 1);
    gemm_wmma<false, false, EPI_BF16><<<gQKV, 256, 0, stream>>>(
        Xb, Dd, 0, Wqb, Dd, 0, Qb, Dd, 0, nullptr, 0, Dd);
    gemm_wmma<false, false, EPI_BF16><<<gQKV, 256, 0, stream>>>(
        Xb, Dd, 0, Wkb, Dd, 0, Kb, Dd, 0, nullptr, 0, Dd);
    gemm_wmma<false, false, EPI_BF16><<<gQKV, 256, 0, stream>>>(
        Xb, Dd, 0, Wvb, Dd, 0, Vb, Dd, 0, nullptr, 0, Dd);

    // 3) Qp/Kp = elu1(Q/K @ proj), projT is [P][D] = [N][K]
    dim3 gFeat(Pp / BN, (int)(M / BM), 1);
    gemm_wmma<false, false, EPI_ELU1><<<gFeat, 256, 0, stream>>>(
        Qb, Dd, 0, projT, Dd, 0, Qp, Pp, 0, nullptr, 0, Dd);
    gemm_wmma<false, false, EPI_ELU1><<<gFeat, 256, 0, stream>>>(
        Kb, Dd, 0, projT, Dd, 0, Kp, Pp, 0, nullptr, 0, Dd);

    // 4) z = 1/(rowsum(Qp)+eps)
    rowsum_inv<<<(int)(M / 8), 256, 0, stream>>>(Qp, zv, Pp);

    // 5) kvT[b] = V[b]^T @ Kp[b]   (A transposed-stage, B [K][N]-stage)
    dim3 gKV(Pp / BN, Dd / BM, Bb);
    gemm_wmma<true, true, EPI_BF16><<<gKV, 256, 0, stream>>>(
        Vb, Dd, (long)Ll * Dd, Kp, Pp, (long)Ll * Pp,
        kvT, Pp, (long)Dd * Pp, nullptr, 0, Ll);

    // 6) out[b] = (Qp[b] @ kvT[b]^T) * z   (kvT is [D][P] = [N][K])
    dim3 gOut(Dd / BN, Ll / BM, Bb);
    gemm_wmma<false, false, EPI_ZOUT><<<gOut, 256, 0, stream>>>(
        Qp, Pp, (long)Ll * Pp, kvT, Pp, (long)Dd * Pp,
        d_out, Dd, (long)Ll * Dd, zv, Ll, Pp);
}
